// Net_orig_14783277432917
// MI455X (gfx1250) — matrix operand order repaired and checked by reading, compiled, not run on hardware
//
#include <hip/hip_runtime.h>

typedef __attribute__((ext_vector_type(16))) __bf16 v16bf;
typedef __attribute__((ext_vector_type(8)))  float  v8f;
typedef __attribute__((ext_vector_type(16))) unsigned short v16us;
typedef __attribute__((ext_vector_type(8)))  unsigned short v8us;

__device__ __forceinline__ unsigned short f2bf(float f) {
    union { float f; unsigned int u; } c; c.f = f;
    unsigned int u = c.u;
    u += 0x7FFFu + ((u >> 16) & 1u);   // round-to-nearest-even
    return (unsigned short)(u >> 16);
}

__device__ __forceinline__ v16us cat8(v8us lo, v8us hi) {
    return __builtin_shufflevector(lo, hi, 0,1,2,3,4,5,6,7,8,9,10,11,12,13,14,15);
}

// ---------------- utility kernels ----------------
__global__ void k_fillf(float* p, long long cnt, float val) {
    long long i = (long long)blockIdx.x * blockDim.x + threadIdx.x;
    long long stride = (long long)gridDim.x * blockDim.x;
    for (; i < cnt; i += stride) p[i] = val;
}

__global__ void k_deg_count(const int* __restrict__ ei, int E, float* __restrict__ deg) {
    int i = blockIdx.x * blockDim.x + threadIdx.x;
    if (i < E) atomicAdd(&deg[ei[(long long)E + i]], 1.0f);   // in-degree on dst
}

__global__ void k_rsqrt(float* p, int n) {
    int i = blockIdx.x * blockDim.x + threadIdx.x;
    if (i < n) p[i] = rsqrtf(p[i]);   // deg >= 1 always (self-loop)
}

// one-time weight conversion:
//   W1T[c][k] = bf16(W1[k][c])   (128 x 512)
//   W2T[c][k] = bf16(W2[k][c])   (48 x 128, cols 40..47 zero-padded)
__global__ void k_prep_w(const float* __restrict__ W1, const float* __restrict__ W2,
                         unsigned short* __restrict__ W1T, unsigned short* __restrict__ W2T) {
    int i = blockIdx.x * blockDim.x + threadIdx.x;
    if (i < 128 * 512) {
        int c = i >> 9, k = i & 511;
        W1T[i] = f2bf(W1[(size_t)k * 128 + c]);
    } else if (i < 128 * 512 + 48 * 128) {
        int j = i - 128 * 512;
        int c = j >> 7, k = j & 127;
        W2T[j] = (c < 40) ? f2bf(W2[(size_t)k * 40 + c]) : (unsigned short)0;
    }
}

// ---------------- GEMM1: h1[n,128] = x[n,512] @ W1[512,128]  (bf16 WMMA, LDS-staged A) ---
__global__ __launch_bounds__(256) void k_gemm1(const float* __restrict__ x,
                                               const unsigned short* __restrict__ W1T,
                                               float* __restrict__ h1, int n) {
    __shared__ unsigned short As[16 * 512];          // 16 KB: one bf16 A row-block
    const int mt = blockIdx.x;

    // cooperative load + convert: 16 rows x 512 cols f32, coalesced float4
    {
        const float4* xb = (const float4*)(x + (size_t)mt * 16 * 512);
        const int t = threadIdx.x;
        #pragma unroll
        for (int j = 0; j < 8; ++j) {
            float4 f = xb[t + 256 * j];
            ushort4 p;
            p.x = f2bf(f.x); p.y = f2bf(f.y); p.z = f2bf(f.z); p.w = f2bf(f.w);
            *(ushort4*)(As + (size_t)(t + 256 * j) * 4) = p;
        }
    }
    __syncthreads();

    const int wave = threadIdx.x >> 5;      // nt = 0..7 (HID/16)
    const int lane = threadIdx.x & 31;
    const int g  = lane >> 4;
    const int ln = lane & 15;
    const int col = wave * 16 + ln;
    const unsigned short* ar = As + (size_t)ln * 512;
    const unsigned short* br = W1T + (size_t)col * 512;

    v8f c0 = {}, c1 = {};
    for (int k0 = 0; k0 < 512; k0 += 64) {
        // tile at k0 -> c0
        {
            v16us ua = cat8(*(const v8us*)(ar + k0 + 8 * g),
                            *(const v8us*)(ar + k0 + 16 + 8 * g));
            v16us ub = cat8(*(const v8us*)(br + k0 + 8 * g),
                            *(const v8us*)(br + k0 + 16 + 8 * g));
            c0 = __builtin_amdgcn_wmma_f32_16x16x32_bf16(
                    false, __builtin_bit_cast(v16bf, ua),
                    false, __builtin_bit_cast(v16bf, ub), (short)0, c0, false, false);
        }
        // tile at k0+32 -> c1 (independent accumulator chain)
        {
            v16us ua = cat8(*(const v8us*)(ar + k0 + 32 + 8 * g),
                            *(const v8us*)(ar + k0 + 48 + 8 * g));
            v16us ub = cat8(*(const v8us*)(br + k0 + 32 + 8 * g),
                            *(const v8us*)(br + k0 + 48 + 8 * g));
            c1 = __builtin_amdgcn_wmma_f32_16x16x32_bf16(
                    false, __builtin_bit_cast(v16bf, ua),
                    false, __builtin_bit_cast(v16bf, ub), (short)0, c1, false, false);
        }
    }
    v8f c = c0 + c1;
    // C/D: vgpr v, lane -> M = v + 8g, N = col  (n is a multiple of 16: no guards)
    float* out = h1 + (size_t)(mt * 16) * 128 + col;
    #pragma unroll
    for (int v = 0; v < 8; ++v) out[(size_t)(v + 8 * g) * 128] = c[v];
}

// ---------------- edge scatter, F=128: agg[dst] += h[src] * dis[src]*dis[dst] ----------------
__global__ __launch_bounds__(256) void k_edge128(const int* __restrict__ ei, int E, int n,
                                                 const float* __restrict__ dis,
                                                 const float* __restrict__ h,
                                                 float* __restrict__ agg) {
    long long e = (long long)blockIdx.x * 8 + (threadIdx.x >> 5);   // one wave per edge
    const int lane = threadIdx.x & 31;
    const long long EA = (long long)E + n;
    if (e >= EA) return;
    int s, d;
    if (e < E) { s = ei[e]; d = ei[(long long)E + e]; }
    else       { s = d = (int)(e - E); }                             // self-loop
    const float nrm = dis[s] * dis[d];
    const float4 v = ((const float4*)(h + (size_t)s * 128))[lane];
    float* o = agg + (size_t)d * 128 + lane * 4;
    atomicAdd(o + 0, v.x * nrm);
    atomicAdd(o + 1, v.y * nrm);
    atomicAdd(o + 2, v.z * nrm);
    atomicAdd(o + 3, v.w * nrm);
}

// ---------------- bias + relu + bf16 pack ----------------
__global__ void k_bias_relu_bf16(const float* __restrict__ agg, const float* __restrict__ b1,
                                 unsigned short* __restrict__ hbf, long long total) {
    long long i = (long long)blockIdx.x * blockDim.x + threadIdx.x;
    if (i >= total) return;
    float f = agg[i] + b1[i & 127];
    hbf[i] = f2bf(fmaxf(f, 0.0f));
}

// ---------------- GEMM2: h2[n,40] = hbf[n,128] @ W2[128,40]  (bf16 WMMA) ----------------
__global__ __launch_bounds__(96) void k_gemm2(const unsigned short* __restrict__ hbf,
                                              const unsigned short* __restrict__ W2T,
                                              float* __restrict__ h2, int n) {
    const int wave = threadIdx.x >> 5;   // nt = 0..2 (cols 0..47, store-guard <40)
    const int lane = threadIdx.x & 31;
    const int mt = blockIdx.x;
    const int g  = lane >> 4;
    const int ln = lane & 15;
    const int col = wave * 16 + ln;
    const unsigned short* ar = hbf + ((size_t)mt * 16 + ln) * 128;
    const unsigned short* br = W2T + (size_t)col * 128;

    v8f c0 = {}, c1 = {};
    #pragma unroll
    for (int k0 = 0; k0 < 128; k0 += 64) {
        {
            v16us ua = cat8(*(const v8us*)(ar + k0 + 8 * g),
                            *(const v8us*)(ar + k0 + 16 + 8 * g));
            v16us ub = cat8(*(const v8us*)(br + k0 + 8 * g),
                            *(const v8us*)(br + k0 + 16 + 8 * g));
            c0 = __builtin_amdgcn_wmma_f32_16x16x32_bf16(
                    false, __builtin_bit_cast(v16bf, ua),
                    false, __builtin_bit_cast(v16bf, ub), (short)0, c0, false, false);
        }
        {
            v16us ua = cat8(*(const v8us*)(ar + k0 + 32 + 8 * g),
                            *(const v8us*)(ar + k0 + 48 + 8 * g));
            v16us ub = cat8(*(const v8us*)(br + k0 + 32 + 8 * g),
                            *(const v8us*)(br + k0 + 48 + 8 * g));
            c1 = __builtin_amdgcn_wmma_f32_16x16x32_bf16(
                    false, __builtin_bit_cast(v16bf, ua),
                    false, __builtin_bit_cast(v16bf, ub), (short)0, c1, false, false);
        }
    }
    v8f c = c0 + c1;
    if (col < 40) {
        float* out = h2 + (size_t)(mt * 16) * 40 + col;
        #pragma unroll
        for (int v = 0; v < 8; ++v) out[(size_t)(v + 8 * g) * 40] = c[v];
    }
}

// ---------------- edge scatter, F=40 ----------------
__global__ __launch_bounds__(256) void k_edge40(const int* __restrict__ ei, int E, int n,
                                                const float* __restrict__ dis,
                                                const float* __restrict__ h,
                                                float* __restrict__ agg) {
    long long e = (long long)blockIdx.x * 8 + (threadIdx.x >> 5);
    const int lane = threadIdx.x & 31;
    const long long EA = (long long)E + n;
    if (e >= EA) return;
    int s, d;
    if (e < E) { s = ei[e]; d = ei[(long long)E + e]; }
    else       { s = d = (int)(e - E); }
    const float nrm = dis[s] * dis[d];
    const float* hr = h + (size_t)s * 40;
    float* o = agg + (size_t)d * 40;
    atomicAdd(o + lane, hr[lane] * nrm);
    if (lane < 8) atomicAdd(o + 32 + lane, hr[32 + lane] * nrm);
}

// ---------------- bias + log_softmax over C=40 ----------------
__global__ void k_log_softmax(const float* __restrict__ agg, const float* __restrict__ b2,
                              float* __restrict__ out, int n) {
    int r = blockIdx.x * blockDim.x + threadIdx.x;
    if (r >= n) return;
    float v[40];
    float m = -3.4e38f;
    #pragma unroll
    for (int c = 0; c < 40; ++c) { v[c] = agg[(size_t)r * 40 + c] + b2[c]; m = fmaxf(m, v[c]); }
    float s = 0.0f;
    #pragma unroll
    for (int c = 0; c < 40; ++c) s += __expf(v[c] - m);
    const float l = m + __logf(s);
    #pragma unroll
    for (int c = 0; c < 40; ++c) out[(size_t)r * 40 + c] = v[c] - l;
}

extern "C" void kernel_launch(void* const* d_in, const int* in_sizes, int n_in,
                              void* d_out, int out_size, void* d_ws, size_t ws_size,
                              hipStream_t stream) {
    const float* x  = (const float*)d_in[0];
    const int*   ei = (const int*)d_in[1];       // [2,E] (src row, dst row)
    const float* W1 = (const float*)d_in[2];     // [512,128]
    const float* b1 = (const float*)d_in[3];     // [128]
    const float* W2 = (const float*)d_in[4];     // [128,40]
    const float* b2 = (const float*)d_in[5];     // [40]
    float* out = (float*)d_out;                  // [n,40]

    const int n = in_sizes[0] / 512;
    const int E = in_sizes[1] / 2;
    const long long EA = (long long)E + n;

    // workspace carve-out (256B aligned), with buffer reuse
    char* ws = (char*)d_ws;
    size_t off = 0;
    auto carve = [&](size_t bytes) -> char* {
        char* p = ws + off; off += (bytes + 255) & ~(size_t)255; return p;
    };
    float*          dis  = (float*)carve((size_t)n * 4);            // deg -> dis (in place)
    float*          h1   = (float*)carve((size_t)n * 128 * 4);
    float*          agg1 = (float*)carve((size_t)n * 128 * 4);
    unsigned short* hbf  = (unsigned short*)carve((size_t)n * 128 * 2);
    unsigned short* W1T  = (unsigned short*)carve((size_t)128 * 512 * 2);
    unsigned short* W2T  = (unsigned short*)carve((size_t)48 * 128 * 2);
    float*          h2   = h1;     // h1 dead after edge128
    float*          agg2 = agg1;   // agg1 dead after bias_relu

    const int mtiles = (n + 15) / 16;     // n divisible by 16 for this workload
    const int eblocks = (int)((EA + 7) / 8);

    // 0) one-time weight transpose+bf16 (tiny)
    k_prep_w<<<(128 * 512 + 48 * 128 + 255) / 256, 256, 0, stream>>>(W1, W2, W1T, W2T);

    // 1) degree (self-loop counts as 1) -> dis = rsqrt(deg)
    k_fillf<<<(n + 255) / 256, 256, 0, stream>>>(dis, n, 1.0f);
    k_deg_count<<<(E + 255) / 256, 256, 0, stream>>>(ei, E, dis);
    k_rsqrt<<<(n + 255) / 256, 256, 0, stream>>>(dis, n);

    // 2) layer 1
    k_gemm1<<<mtiles, 256, 0, stream>>>(x, W1T, h1, n);
    k_fillf<<<4096, 256, 0, stream>>>(agg1, (long long)n * 128, 0.0f);
    k_edge128<<<eblocks, 256, 0, stream>>>(ei, E, n, dis, h1, agg1);
    k_bias_relu_bf16<<<(int)(((long long)n * 128 + 255) / 256), 256, 0, stream>>>(
        agg1, b1, hbf, (long long)n * 128);

    // 3) layer 2
    k_gemm2<<<mtiles, 96, 0, stream>>>(hbf, W2T, h2, n);
    k_fillf<<<4096, 256, 0, stream>>>(agg2, (long long)n * 40, 0.0f);
    k_edge40<<<eblocks, 256, 0, stream>>>(ei, E, n, dis, h2, agg2);

    // 4) log_softmax
    k_log_softmax<<<(n + 255) / 256, 256, 0, stream>>>(agg2, b2, out, n);
}
